// RNN_39719857553990
// MI455X (gfx1250) — compile-verified, matrix operand order checked by reading, not run on hardware
//
#include <hip/hip_runtime.h>
#include <hip/hip_bf16.h>
#include <math.h>

typedef __attribute__((ext_vector_type(16))) __bf16 v16bf;
typedef __attribute__((ext_vector_type(8)))  __bf16 v8bf;
typedef __attribute__((ext_vector_type(4)))  __bf16 v4bf;
typedef __attribute__((ext_vector_type(8)))  float  v8f;
typedef __attribute__((ext_vector_type(4)))  float  v4f;

#define B_ 256
#define T_ 512
#define F_ 512
#define H_ 512
#define NG 2048   // 4*H
#define K_ 1024   // F+H

union Frag { v16bf v; v8bf h[2]; };

__device__ __forceinline__ void ldfrag(Frag& f, const __bf16* p) {
  f.h[0] = *(const v8bf*)p;
  f.h[1] = *(const v8bf*)(p + 16);
}

__device__ __forceinline__ v8f wmma_bf16(const Frag& a, const Frag& b, v8f c) {
  return __builtin_amdgcn_wmma_f32_16x16x32_bf16(
      false, a.v, false, b.v, (short)0, c, false, false);
}

__device__ __forceinline__ float sigm(float v) {
  return 1.0f / (1.0f + __expf(-v));
}

// ---------------------------------------------------------------------------
// Prep: Wt[n][k] = bf16(W[k][n]) (transposed, coalesced writes), zero h0 & c.
// ---------------------------------------------------------------------------
__global__ __launch_bounds__(256)
void lstm_prep(const float* __restrict__ W, __bf16* __restrict__ Wt,
               __bf16* __restrict__ h0, float* __restrict__ c) {
  int idx = blockIdx.x * blockDim.x + threadIdx.x;
  if (idx < K_ * NG) {
    int n = idx / K_;
    int k = idx - n * K_;
    Wt[idx] = (__bf16)W[(size_t)k * NG + n];
  }
  if (idx < B_ * H_) {
    h0[idx] = (__bf16)0.0f;
    c[idx]  = 0.0f;
  }
}

// x (f32) -> xbf (bf16), 4 elements per thread, coalesced.
__global__ __launch_bounds__(256)
void lstm_xbf(const float* __restrict__ x, __bf16* __restrict__ xbf) {
  size_t i = ((size_t)blockIdx.x * 256 + threadIdx.x) * 4;
  v4f v = *(const v4f*)(x + i);
  v4bf o;
#pragma unroll
  for (int e = 0; e < 4; ++e) o[e] = (__bf16)v[e];
  *(v4bf*)(xbf + i) = o;
}

// ---------------------------------------------------------------------------
// Phase 1 (parallel): XW[m][n] = xbf[m][:512] @ Wx + bias,  m in [0, B*T).
// Block = 4 waves; wave w owns M-subtile m0+16w and computes 4 N-subtiles.
// A/B fragments are double-buffered so loads overlap WMMA issue.
// ---------------------------------------------------------------------------
__global__ __launch_bounds__(128)
void lstm_xw(const __bf16* __restrict__ xbf, const __bf16* __restrict__ Wt,
             const float* __restrict__ bias, float* __restrict__ XW) {
  const int lane = threadIdx.x & 31;
  const int wv   = threadIdx.x >> 5;
  const int nrow = lane & 15;
  const bool hi  = lane >= 16;
  const int koff = hi ? 8 : 0;
  const size_t m0 = (size_t)blockIdx.y * 64 + (size_t)wv * 16;
  const int n0 = blockIdx.x * 64;

  const __bf16* arow = xbf + (m0 + nrow) * F_;
  const __bf16* w0   = Wt + (size_t)(n0 + nrow) * K_;  // x-part: k in [0,512)

  v8f acc0 = {}, acc1 = {}, acc2 = {}, acc3 = {};
  Frag A0, A1, Bf0[4], Bf1[4];

  // prime buffer 0 with kb = 0
  ldfrag(A0, arow + koff);
#pragma unroll
  for (int j = 0; j < 4; ++j) ldfrag(Bf0[j], w0 + (size_t)j * 16 * K_ + koff);

  for (int kb = 0; kb < F_ / 32; kb += 2) {
    // prefetch kb+1 into buffer 1 (kb+1 < 16 always: 16 k-blocks, 8 iterations)
    {
      const int kn = (kb + 1) * 32 + koff;
      ldfrag(A1, arow + kn);
#pragma unroll
      for (int j = 0; j < 4; ++j) ldfrag(Bf1[j], w0 + (size_t)j * 16 * K_ + kn);
    }
    // compute kb from buffer 0
    acc0 = wmma_bf16(A0, Bf0[0], acc0);
    acc1 = wmma_bf16(A0, Bf0[1], acc1);
    acc2 = wmma_bf16(A0, Bf0[2], acc2);
    acc3 = wmma_bf16(A0, Bf0[3], acc3);
    // prefetch kb+2 into buffer 0
    if (kb + 2 < F_ / 32) {
      const int kn = (kb + 2) * 32 + koff;
      ldfrag(A0, arow + kn);
#pragma unroll
      for (int j = 0; j < 4; ++j) ldfrag(Bf0[j], w0 + (size_t)j * 16 * K_ + kn);
    }
    // compute kb+1 from buffer 1
    acc0 = wmma_bf16(A1, Bf1[0], acc0);
    acc1 = wmma_bf16(A1, Bf1[1], acc1);
    acc2 = wmma_bf16(A1, Bf1[2], acc2);
    acc3 = wmma_bf16(A1, Bf1[3], acc3);
  }

  const int mb = hi ? 8 : 0;
#pragma unroll
  for (int j = 0; j < 4; ++j) {
    v8f a = (j == 0) ? acc0 : (j == 1) ? acc1 : (j == 2) ? acc2 : acc3;
    const int col = n0 + j * 16 + nrow;
    const float bv = bias[col];
#pragma unroll
    for (int v = 0; v < 8; ++v)
      XW[(m0 + mb + v) * NG + col] = a[v] + bv;
  }
}

// ---------------------------------------------------------------------------
// Fast recurrent step: gates = XW[:,t,:] + h @ Wh.  K = 512, double-buffered.
// ---------------------------------------------------------------------------
__global__ __launch_bounds__(128)
void lstm_step_fast(const float* __restrict__ XW, const __bf16* __restrict__ Wt,
                    const __bf16* __restrict__ hin, __bf16* __restrict__ hout,
                    float* __restrict__ c, float* __restrict__ out, int t) {
  const int lane = threadIdx.x & 31;
  const int g    = threadIdx.x >> 5;     // gate: 0=i 1=j 2=f 3=o
  const int nrow = lane & 15;
  const bool hi  = lane >= 16;
  const int koff = hi ? 8 : 0;
  const int m0   = blockIdx.y * 16;      // batch tile origin
  const int n0   = blockIdx.x * 16;      // hidden tile origin

  __shared__ float lds_g[4][16][16];

  const __bf16* hrow = hin + (size_t)(m0 + nrow) * H_;
  const __bf16* wrow = Wt + (size_t)(g * H_ + n0 + nrow) * K_ + F_;  // h-part

  v8f acc = {};
  Frag A0, A1, Bf0, Bf1;
  ldfrag(A0, hrow + koff);
  ldfrag(Bf0, wrow + koff);

  for (int kb = 0; kb < H_ / 32; kb += 2) {
    {
      const int kn = (kb + 1) * 32 + koff;
      ldfrag(A1, hrow + kn);
      ldfrag(Bf1, wrow + kn);
    }
    acc = wmma_bf16(A0, Bf0, acc);
    if (kb + 2 < H_ / 32) {
      const int kn = (kb + 2) * 32 + koff;
      ldfrag(A0, hrow + kn);
      ldfrag(Bf0, wrow + kn);
    }
    acc = wmma_bf16(A1, Bf1, acc);
  }

  const int mbase = hi ? 8 : 0;
#pragma unroll
  for (int v = 0; v < 8; ++v)
    lds_g[g][mbase + v][nrow] = acc[v];
  __syncthreads();

#pragma unroll
  for (int e = threadIdx.x; e < 256; e += 128) {
    const int M = e >> 4, N = e & 15;
    const size_t xwrow = ((size_t)(m0 + M) * T_ + t) * NG + n0 + N;
    const float iv = lds_g[0][M][N] + XW[xwrow];
    const float jv = lds_g[1][M][N] + XW[xwrow + H_];
    const float fv = lds_g[2][M][N] + XW[xwrow + 2 * H_];
    const float ov = lds_g[3][M][N] + XW[xwrow + 3 * H_];
    const size_t cidx = (size_t)(m0 + M) * H_ + (n0 + N);
    const float cold = c[cidx];
    const float cn = cold * sigm(fv + 1.0f) + sigm(iv) * tanhf(jv);
    const float hn = tanhf(cn) * sigm(ov);
    c[cidx] = cn;
    hout[cidx] = (__bf16)hn;
    out[((size_t)(m0 + M) * T_ + t) * H_ + (n0 + N)] = hn;
  }
}

// ---------------------------------------------------------------------------
// Fallback fused step (small workspace): gates = [x_t | h] @ W + b, K = 1024.
// ---------------------------------------------------------------------------
__global__ __launch_bounds__(128)
void lstm_step(const float* __restrict__ x, const __bf16* __restrict__ Wt,
               const float* __restrict__ bias, const __bf16* __restrict__ hin,
               __bf16* __restrict__ hout, float* __restrict__ c,
               float* __restrict__ out, int t) {
  const int lane = threadIdx.x & 31;
  const int g    = threadIdx.x >> 5;
  const int nrow = lane & 15;
  const bool hi  = lane >= 16;
  const int koff = hi ? 8 : 0;
  const int m0   = blockIdx.y * 16;
  const int n0   = blockIdx.x * 16;

  __shared__ float lds_g[4][16][16];

  const float*  xrow = x   + ((size_t)(m0 + nrow) * T_ + t) * F_;
  const __bf16* hrow = hin + (size_t)(m0 + nrow) * H_;
  const __bf16* wrow = Wt  + (size_t)(g * H_ + n0 + nrow) * K_;

  v8f acc = {};
  for (int kb = 0; kb < F_ / 32; ++kb) {
    const int k0 = kb * 32 + koff;
    v4f x0 = *(const v4f*)(xrow + k0);
    v4f x1 = *(const v4f*)(xrow + k0 + 4);
    v4f x2 = *(const v4f*)(xrow + k0 + 16);
    v4f x3 = *(const v4f*)(xrow + k0 + 20);
    Frag a;
#pragma unroll
    for (int e = 0; e < 4; ++e) {
      a.v[e]      = (__bf16)x0[e];
      a.v[e + 4]  = (__bf16)x1[e];
      a.v[e + 8]  = (__bf16)x2[e];
      a.v[e + 12] = (__bf16)x3[e];
    }
    Frag Bf;
    ldfrag(Bf, wrow + k0);
    acc = wmma_bf16(a, Bf, acc);
  }
  for (int kb = F_ / 32; kb < K_ / 32; ++kb) {
    const int k0 = kb * 32 + koff;
    Frag A, Bf;
    ldfrag(A, hrow + (k0 - F_));
    ldfrag(Bf, wrow + k0);
    acc = wmma_bf16(A, Bf, acc);
  }

  const float bv = bias[g * H_ + n0 + nrow];
  const int mbase = hi ? 8 : 0;
#pragma unroll
  for (int v = 0; v < 8; ++v)
    lds_g[g][mbase + v][nrow] = acc[v] + bv;
  __syncthreads();

#pragma unroll
  for (int e = threadIdx.x; e < 256; e += 128) {
    const int M = e >> 4, N = e & 15;
    const float iv = lds_g[0][M][N];
    const float jv = lds_g[1][M][N];
    const float fv = lds_g[2][M][N];
    const float ov = lds_g[3][M][N];
    const size_t cidx = (size_t)(m0 + M) * H_ + (n0 + N);
    const float cold = c[cidx];
    const float cn = cold * sigm(fv + 1.0f) + sigm(iv) * tanhf(jv);
    const float hn = tanhf(cn) * sigm(ov);
    c[cidx] = cn;
    hout[cidx] = (__bf16)hn;
    out[((size_t)(m0 + M) * T_ + t) * H_ + (n0 + N)] = hn;
  }
}

// ---------------------------------------------------------------------------
extern "C" void kernel_launch(void* const* d_in, const int* in_sizes, int n_in,
                              void* d_out, int out_size, void* d_ws, size_t ws_size,
                              hipStream_t stream) {
  const float* x    = (const float*)d_in[0];   // [B, T, F]
  const float* W    = (const float*)d_in[1];   // [F+H, 4H]
  const float* bias = (const float*)d_in[2];   // [4H]
  float* out = (float*)d_out;                  // [B, T, H]
  char* ws = (char*)d_ws;

  const size_t WT_B  = (size_t)NG * K_ * 2;          // 4 MiB
  const size_t H_B   = (size_t)B_ * H_ * 2;          // 256 KiB each
  const size_t C_B   = (size_t)B_ * H_ * 4;          // 512 KiB
  const size_t XBF_B = (size_t)B_ * T_ * F_ * 2;     // 128 MiB
  const size_t XW_B  = (size_t)B_ * T_ * NG * 4;     // 1 GiB

  __bf16* Wt = (__bf16*)ws;
  __bf16* hA = (__bf16*)(ws + WT_B);
  __bf16* hB = (__bf16*)(ws + WT_B + H_B);
  float*  c  = (float*)(ws + WT_B + 2 * H_B);
  __bf16* xbf = (__bf16*)(ws + WT_B + 2 * H_B + C_B);
  float*  XW  = (float*)(ws + WT_B + 2 * H_B + C_B + XBF_B);

  const size_t FAST_WS = WT_B + 2 * H_B + C_B + XBF_B + XW_B;

  lstm_prep<<<(K_ * NG + 255) / 256, 256, 0, stream>>>(W, Wt, hA, c);

  if (ws_size >= FAST_WS) {
    // Split path: parallel input GEMM, then K=512 recurrence.
    lstm_xbf<<<(int)((size_t)B_ * T_ * F_ / 4 / 256), 256, 0, stream>>>(x, xbf);
    lstm_xw<<<dim3(NG / 64, (B_ * T_) / 64), 128, 0, stream>>>(xbf, Wt, bias, XW);
    for (int t = 0; t < T_; ++t) {
      const __bf16* hin = (t & 1) ? hB : hA;
      __bf16* hout      = (t & 1) ? hA : hB;
      lstm_step_fast<<<dim3(H_ / 16, B_ / 16), 128, 0, stream>>>(
          XW, Wt, hin, hout, c, out, t);
    }
  } else {
    // Fused fallback (needs only ~5 MiB of workspace).
    for (int t = 0; t < T_; ++t) {
      const __bf16* hin = (t & 1) ? hB : hA;
      __bf16* hout      = (t & 1) ? hA : hB;
      lstm_step<<<dim3(H_ / 16, B_ / 16), 128, 0, stream>>>(
          x, Wt, bias, hin, hout, c, out, t);
    }
  }
}
